// DGNNLayer_24051816858240
// MI455X (gfx1250) — matrix operand order repaired and checked by reading, not compiled, not run on hardware
//
#include <hip/hip_runtime.h>
#include <hip/hip_bf16.h>

typedef __attribute__((ext_vector_type(16))) _Float16 v16h;
typedef __attribute__((ext_vector_type(8)))  _Float16 v8h;
typedef __attribute__((ext_vector_type(8)))  float    v8f;

#define WMMA_F16(a, b, c) \
  __builtin_amdgcn_wmma_f32_16x16x32_f16(false, (a), false, (b), (short)0, (c), false, false)

// ---------- WMMA fragment loaders (contiguous k order, same for A and B) ----------
// A 16x32 (MxK): lanes 0-15 -> M=lane, K = 0..15 ; lanes 16-31 -> M=lane-16, K = 16..31
// p is 16B aligned -> two ds_load_b128
__device__ __forceinline__ v16h load_a_lds(const _Float16* base, int stride, int row, int k0, int lane) {
  const _Float16* p = base + row * stride + k0 + ((lane >> 4) * 16);
  v8h lo = *(const v8h*)(p);
  v8h hi = *(const v8h*)(p + 8);
  return __builtin_shufflevector(lo, hi, 0, 1, 2, 3, 4, 5, 6, 7, 8, 9, 10, 11, 12, 13, 14, 15);
}
// B 32x16 (KxN), weights pre-transposed Wt[n][k]: lane's column n holds 16 contiguous K values
// p is 32B aligned -> clause of two global_load_b128
__device__ __forceinline__ v16h load_b_row(const _Float16* w, int Kdim, int n, int k0, int lane) {
  const _Float16* p = w + (size_t)n * Kdim + k0 + ((lane >> 4) * 16);
  return *(const v16h*)p;
}

// ---------- monotonic float<->uint encoding for atomic max ----------
__device__ __forceinline__ unsigned fenc(float f) {
  unsigned u = __float_as_uint(f);
  return (u & 0x80000000u) ? ~u : (u | 0x80000000u);
}
__device__ __forceinline__ float fdec(unsigned u) {
  return __uint_as_float((u & 0x80000000u) ? (u ^ 0x80000000u) : ~u);
}
#define NEGINF_ENC 0x007FFFFFu  // fenc(-inf)

// ---------- weight conversion: f32 -> f16, transposed to [n][k] ----------
__global__ void conv_w_kernel(const float* __restrict__ Wq, const float* __restrict__ Wk,
                              const float* __restrict__ Wv, const float* __restrict__ W1,
                              const float* __restrict__ W2, _Float16* __restrict__ w16) {
  int i = blockIdx.x * 256 + threadIdx.x;
  if (i < 3 * 16384) {                        // Wq|Wk|Wv transposed [128][128]
    int mat = i / 16384, r = i % 16384, n = r >> 7, k = r & 127;
    const float* W = (mat == 0) ? Wq : ((mat == 1) ? Wk : Wv);
    w16[i] = (_Float16)W[k * 128 + n];
  } else if (i < 3 * 16384 + 32768) {         // W1t [256][128]
    int j = i - 3 * 16384, n = j >> 7, k = j & 127;
    w16[i] = (_Float16)W1[k * 256 + n];
  } else if (i < 3 * 16384 + 32768 + 32768) { // W2t [128][256]
    int j = i - 3 * 16384 - 32768, n = j >> 8, k = j & 255;
    w16[i] = (_Float16)W2[k * 128 + n];
  }
}

// ---------- fused QKV projection: [R x 128] @ [128 x 128] + bias, R = T*N ----------
__global__ void qkv_gemm_kernel(const float* __restrict__ x, const _Float16* __restrict__ w16,
                                const float* __restrict__ bq, const float* __restrict__ bk,
                                const float* __restrict__ bv, float* __restrict__ qkv, int R) {
  __shared__ __align__(16) _Float16 Xs[64 * 128];
  const int tid = threadIdx.x;
  const int rowbase = blockIdx.x * 64;
  const int mat = blockIdx.y;
  for (int i = tid; i < 64 * 128; i += 256) {
    int r = i >> 7, cc = i & 127, gr = rowbase + r;
    Xs[i] = (gr < R) ? (_Float16)x[(size_t)gr * 128 + cc] : (_Float16)0.0f;
  }
  __syncthreads();
  const _Float16* W = w16 + mat * 16384;
  const float* bias = (mat == 0) ? bq : ((mat == 1) ? bk : bv);
  float* out = qkv + (size_t)mat * R * 128;
  const int wave = tid >> 5, lane = tid & 31, hsel = lane >> 4, l15 = lane & 15;
  const bool full = (rowbase + 64 <= R);  // block-uniform
  for (int tile = wave; tile < 32; tile += 8) {  // 4 row-tiles x 8 col-tiles
    int rt = tile & 3, ct = tile >> 2;
    v8f c = {};
#pragma unroll
    for (int kk = 0; kk < 4; ++kk) {
      v16h a = load_a_lds(Xs, 128, rt * 16 + l15, kk * 32, lane);
      v16h b = load_b_row(W, 128, ct * 16 + l15, kk * 32, lane);
      c = WMMA_F16(a, b, c);
    }
    int col = ct * 16 + l15;
    float bb = bias[col];
    float* orow = out + (size_t)(rowbase + rt * 16 + hsel * 8) * 128 + col;
    if (full) {
#pragma unroll
      for (int r = 0; r < 8; ++r) orow[(size_t)r * 128] = c[r] + bb;
    } else {
#pragma unroll
      for (int r = 0; r < 8; ++r) {
        int grow = rowbase + rt * 16 + hsel * 8 + r;
        if (grow < R) out[(size_t)grow * 128 + col] = c[r] + bb;
      }
    }
  }
}

// ---------- per-t_tar accumulator init ----------
__global__ void att_init_kernel(unsigned* rmax, unsigned* smax, float* rden, float* sden,
                                float* racc, float* sacc, int N) {
  int i = blockIdx.x * 256 + threadIdx.x;
  if (i < N * 128) { racc[i] = 0.f; sacc[i] = 0.f; }
  if (i < N * 8)   { rden[i] = 0.f; sden[i] = 0.f; rmax[i] = NEGINF_ENC; smax[i] = NEGINF_ENC; }
}

// ---------- edge pass A: att and per-(node,head) max (for softmax stability) ----------
__global__ void edge_max_kernel(const long long* __restrict__ ei, const float* __restrict__ Q,
                                const float* __restrict__ K, unsigned* __restrict__ rmax,
                                unsigned* __restrict__ smax, int E, int nthreads, int t_tar, int N) {
  int gid = blockIdx.x * 256 + threadIdx.x;
  if (gid >= nthreads) return;
  int e = gid >> 3, h = gid & 7;
  int ts = e / E, idx = e - ts * E;
  const long long* base = ei + (size_t)ts * 2 * E;
  int src = (int)base[idx];
  int tgt = (int)base[E + idx];
  const float* q = Q + ((size_t)t_tar * N + tgt) * 128 + h * 16;
  const float* k = K + ((size_t)ts * N + src) * 128 + h * 16;
  float d = 0.f;
#pragma unroll
  for (int j = 0; j < 16; ++j) d += q[j] * k[j];
  float att = d * 0.25f;  // 1/sqrt(dk=16)
  atomicMax(&rmax[tgt * 8 + h], fenc(att));
  atomicMax(&smax[tgt * 8 + h], fenc(-att));
}

// ---------- edge pass B: exp weights, denominators, weighted-V accumulation ----------
__global__ void edge_accum_kernel(const long long* __restrict__ ei, const float* __restrict__ Q,
                                  const float* __restrict__ K, const float* __restrict__ V,
                                  const unsigned* __restrict__ rmax, const unsigned* __restrict__ smax,
                                  float* __restrict__ rden, float* __restrict__ sden,
                                  float* __restrict__ racc, float* __restrict__ sacc,
                                  int E, int nthreads, int t_tar, int N) {
  int gid = blockIdx.x * 256 + threadIdx.x;
  if (gid >= nthreads) return;
  int e = gid >> 3, h = gid & 7;
  int ts = e / E, idx = e - ts * E;
  const long long* base = ei + (size_t)ts * 2 * E;
  int src = (int)base[idx];
  int tgt = (int)base[E + idx];
  const float* q = Q + ((size_t)t_tar * N + tgt) * 128 + h * 16;
  const float* k = K + ((size_t)ts * N + src) * 128 + h * 16;
  float d = 0.f;
#pragma unroll
  for (int j = 0; j < 16; ++j) d += q[j] * k[j];
  float att = d * 0.25f;
  float er = __expf(att  - fdec(rmax[tgt * 8 + h]));
  float es = __expf(-att - fdec(smax[tgt * 8 + h]));
  atomicAdd(&rden[tgt * 8 + h], er);
  atomicAdd(&sden[tgt * 8 + h], es);
  const float* v = V + ((size_t)ts * N + src) * 128 + h * 16;
  float* ra = racc + (size_t)tgt * 128 + h * 16;
  float* sa = sacc + (size_t)tgt * 128 + h * 16;
#pragma unroll
  for (int j = 0; j < 16; ++j) { atomicAdd(&ra[j], er * v[j]); atomicAdd(&sa[j], es * v[j]); }
}

// ---------- node finalize + FFN (LN -> W1+gelu -> W2 + residual), WMMA ----------
__device__ __forceinline__ float hat_val(const float* acc, const float* den, const float* xr,
                                         int addx, int node, int j) {
  float a = acc[(size_t)node * 128 + j] / (den[node * 8 + (j >> 4)] + 1e-16f);
  if (addx) a += xr[(size_t)node * 128 + j];
  return a;
}

__global__ void ffn_kernel(const float* __restrict__ racc, const float* __restrict__ rden,
                           const float* __restrict__ sacc, const float* __restrict__ sden,
                           const float* __restrict__ xt, const float* __restrict__ ln_g,
                           const float* __restrict__ ln_b, const _Float16* __restrict__ w1t,
                           const float* __restrict__ b1, const _Float16* __restrict__ w2t,
                           const float* __restrict__ b2, float* __restrict__ out_c,
                           float* __restrict__ out_s, int N) {
  __shared__ __align__(16) _Float16 Hs[64 * 128];  // LN output (f16)
  __shared__ __align__(16) _Float16 Gs[64 * 256];  // gelu(H@W1+b1) (f16)
  const int tid = threadIdx.x;
  const int rowbase = blockIdx.x * 64;
  const int causal = (blockIdx.y == 0);
  const float* acc = causal ? racc : sacc;
  const float* den = causal ? rden : sden;
  float* out = causal ? out_c : out_s;

  if (tid < 64) {
    int node = rowbase + tid;
    if (node < N) {
      float s1 = 0.f, s2 = 0.f;
      for (int j = 0; j < 128; ++j) {
        float hv = hat_val(acc, den, xt, causal, node, j);
        s1 += hv; s2 += hv * hv;
      }
      float mu = s1 * (1.0f / 128.0f);
      float var = s2 * (1.0f / 128.0f) - mu * mu;
      float rstd = rsqrtf(var + 1e-5f);
      for (int j = 0; j < 128; ++j) {
        float hv = hat_val(acc, den, xt, causal, node, j);
        Hs[tid * 128 + j] = (_Float16)(((hv - mu) * rstd) * ln_g[j] + ln_b[j]);
      }
    } else {
      for (int j = 0; j < 128; ++j) Hs[tid * 128 + j] = (_Float16)0.0f;
    }
  }
  __syncthreads();

  const int wave = tid >> 5, lane = tid & 31, hsel = lane >> 4, l15 = lane & 15;
  const bool full = (rowbase + 64 <= N);  // block-uniform

  // GEMM1: [64x128] @ [128x256] + b1, exact gelu -> Gs
  for (int tile = wave; tile < 64; tile += 8) {  // 4 row-tiles x 16 col-tiles
    int rt = tile & 3, ct = tile >> 2;
    v8f c = {};
#pragma unroll
    for (int kk = 0; kk < 4; ++kk) {
      v16h a = load_a_lds(Hs, 128, rt * 16 + l15, kk * 32, lane);
      v16h b = load_b_row(w1t, 128, ct * 16 + l15, kk * 32, lane);
      c = WMMA_F16(a, b, c);
    }
    int col = ct * 16 + l15;
    float bb = b1[col];
#pragma unroll
    for (int r = 0; r < 8; ++r) {
      float v = c[r] + bb;
      float g = 0.5f * v * (1.0f + erff(v * 0.70710678118f));
      Gs[(rt * 16 + hsel * 8 + r) * 256 + col] = (_Float16)g;
    }
  }
  __syncthreads();

  // GEMM2: [64x256] @ [256x128] + b2 + h (residual) -> out
  for (int tile = wave; tile < 32; tile += 8) {  // 4 row-tiles x 8 col-tiles
    int rt = tile & 3, ct = tile >> 2;
    v8f c = {};
#pragma unroll
    for (int kk = 0; kk < 8; ++kk) {
      v16h a = load_a_lds(Gs, 256, rt * 16 + l15, kk * 32, lane);
      v16h b = load_b_row(w2t, 256, ct * 16 + l15, kk * 32, lane);
      c = WMMA_F16(a, b, c);
    }
    int col = ct * 16 + l15;
    float bb = b2[col];
    int mrow = rt * 16 + hsel * 8;
    if (full) {
#pragma unroll
      for (int r = 0; r < 8; ++r) {
        int node = rowbase + mrow + r;
        float hv = hat_val(acc, den, xt, causal, node, col);
        out[(size_t)node * 128 + col] = c[r] + bb + hv;
      }
    } else {
#pragma unroll
      for (int r = 0; r < 8; ++r) {
        int node = rowbase + mrow + r;
        if (node < N) {
          float hv = hat_val(acc, den, xt, causal, node, col);
          out[(size_t)node * 128 + col] = c[r] + bb + hv;
        }
      }
    }
  }
}

__global__ void add_kernel(const float* __restrict__ a, const float* __restrict__ b,
                           float* __restrict__ o, int n) {
  int i = blockIdx.x * 256 + threadIdx.x;
  if (i < n) o[i] = a[i] + b[i];
}

extern "C" void kernel_launch(void* const* d_in, const int* in_sizes, int n_in,
                              void* d_out, int out_size, void* d_ws, size_t ws_size,
                              hipStream_t stream) {
  const float*     x    = (const float*)d_in[0];
  const long long* ei   = (const long long*)d_in[1];
  const float*     Wq   = (const float*)d_in[2];
  const float*     bq   = (const float*)d_in[3];
  const float*     Wk   = (const float*)d_in[4];
  const float*     bk   = (const float*)d_in[5];
  const float*     Wv   = (const float*)d_in[6];
  const float*     bv   = (const float*)d_in[7];
  const float*     ln_g = (const float*)d_in[8];
  const float*     ln_b = (const float*)d_in[9];
  const float*     W1   = (const float*)d_in[10];
  const float*     b1   = (const float*)d_in[11];
  const float*     W2   = (const float*)d_in[12];
  const float*     b2   = (const float*)d_in[13];

  const int T = 3;
  const int N = in_sizes[0] / (T * 128);
  const int E = in_sizes[1] / (T * 2);
  const int R = T * N;
  const size_t NT = (size_t)N * 128;

  // workspace layout
  float*    Q    = (float*)d_ws;                 // [T*N,128]
  float*    Kp   = Q  + (size_t)R * 128;         // [T*N,128]
  float*    Vp   = Kp + (size_t)R * 128;         // [T*N,128]
  float*    rden = Vp + (size_t)R * 128;         // [N,8]
  float*    sden = rden + (size_t)N * 8;
  unsigned* rmax = (unsigned*)(sden + (size_t)N * 8);
  unsigned* smax = rmax + (size_t)N * 8;
  float*    racc = (float*)(smax + (size_t)N * 8);  // [N,128]
  float*    sacc = racc + NT;                       // [N,128]
  _Float16* w16  = (_Float16*)(sacc + NT);          // f16 weights
  const _Float16* w1t = w16 + 3 * 16384;            // [256][128]
  const _Float16* w2t = w1t + 256 * 128;            // [128][256]

  float* xs_out = (float*)d_out;
  float* cs_out = xs_out + (size_t)T * NT;
  float* ss_out = cs_out + (size_t)T * NT;

  conv_w_kernel<<<(114688 + 255) / 256, 256, 0, stream>>>(Wq, Wk, Wv, W1, W2, w16);

  dim3 gq((R + 63) / 64, 3);
  qkv_gemm_kernel<<<gq, 256, 0, stream>>>(x, w16, bq, bk, bv, Q, R);

  for (int t = 0; t < T; ++t) {
    att_init_kernel<<<((int)NT + 255) / 256, 256, 0, stream>>>(rmax, smax, rden, sden, racc, sacc, N);
    int nth = (t + 1) * E * 8;
    edge_max_kernel<<<(nth + 255) / 256, 256, 0, stream>>>(ei, Q, Kp, rmax, smax, E, nth, t, N);
    edge_accum_kernel<<<(nth + 255) / 256, 256, 0, stream>>>(ei, Q, Kp, Vp, rmax, smax,
                                                             rden, sden, racc, sacc, E, nth, t, N);
    dim3 gf((N + 63) / 64, 2);
    ffn_kernel<<<gf, 256, 0, stream>>>(racc, rden, sacc, sden, x + (size_t)t * NT, ln_g, ln_b,
                                       w1t, b1, w2t, b2,
                                       cs_out + (size_t)t * NT, ss_out + (size_t)t * NT, N);
    add_kernel<<<((int)NT + 255) / 256, 256, 0, stream>>>(cs_out + (size_t)t * NT,
                                                          ss_out + (size_t)t * NT,
                                                          xs_out + (size_t)t * NT, (int)NT);
  }
}
